// Block_68221260529679
// MI455X (gfx1250) — matrix operand order repaired and checked by reading, not compiled, not run on hardware
//
#include <hip/hip_runtime.h>
#include <hip/hip_bf16.h>
#include <math.h>

typedef __bf16 bf16_t;
typedef __attribute__((ext_vector_type(16))) __bf16 v16bf;
typedef __attribute__((ext_vector_type(8)))  float  v8f;

// Problem dims (fixed by the reference)
constexpr int kB = 2, kT = 2048, kD = 2048, kN = 8, kH = 256, kF = 16384;
constexpr long long kMT = (long long)kB * kT;      // 4096 rows
constexpr long long kNH = (long long)kN * kH;      // 2048

// GEMM tiling
constexpr int TM = 128, TN = 128, TK = 32;
constexpr int LDSS = 40;  // bf16 elements per LDS row (32 + 8 pad), 80B row stride

// Async Global->LDS copy of 16B per lane; INST_OFFSET applies to both LDS and
// global addresses (ISA 08_async_tensor §4.4), tracked by ASYNCcnt.
__device__ __forceinline__ void async_cp16(unsigned lds_addr, const void* gaddr) {
  asm volatile("global_load_async_to_lds_b128 %0, %1, off"
               :: "v"(lds_addr), "v"(gaddr) : "memory");
}
__device__ __forceinline__ void async_cp16_off16(unsigned lds_addr, const void* gaddr) {
  asm volatile("global_load_async_to_lds_b128 %0, %1, off offset:16"
               :: "v"(lds_addr), "v"(gaddr) : "memory");
}
__device__ __forceinline__ void wait_async0() {
  asm volatile("s_wait_asynccnt 0x0" ::: "memory");
}
__device__ __forceinline__ unsigned lds_off(const void* p) {
  // generic address of __shared__ object: low 32 bits == LDS byte address
  return (unsigned)(unsigned long long)p;
}

// ---------------------------------------------------------------- elementwise
__global__ void cvt_f32_bf16(const float* __restrict__ in, bf16_t* __restrict__ out,
                             long long n) {
  long long i = (long long)blockIdx.x * blockDim.x + threadIdx.x;
  long long stride = (long long)gridDim.x * blockDim.x;
  for (; i < n; i += stride) out[i] = (bf16_t)in[i];
}

// w_q (N,D,H) f32  ->  (D, N*H) bf16 packed for GEMM B operand
__global__ void pack_wq(const float* __restrict__ wq, bf16_t* __restrict__ out) {
  long long i = (long long)blockIdx.x * blockDim.x + threadIdx.x;  // over D*N*H outputs
  long long total = (long long)kD * kN * kH;
  if (i >= total) return;
  int h = (int)(i % kH);
  long long t = i / kH;
  int n = (int)(t % kN);
  int d = (int)(t / kN);
  out[i] = (bf16_t)wq[((long long)n * kD + d) * kH + h];
}

// One block per row of D=2048; out = x * rsqrt(mean(x^2)+eps) * (1+scale), bf16
__global__ void rmsnorm_bf16(const float* __restrict__ x, const float* __restrict__ scale,
                             bf16_t* __restrict__ out) {
  __shared__ float red[256];
  long long row = blockIdx.x;
  const float* xr = x + row * kD;
  float v[8];
  float s = 0.f;
#pragma unroll
  for (int j = 0; j < 8; ++j) { v[j] = xr[threadIdx.x + 256 * j]; s += v[j] * v[j]; }
  red[threadIdx.x] = s; __syncthreads();
  for (int o = 128; o > 0; o >>= 1) {
    if ((int)threadIdx.x < o) red[threadIdx.x] += red[threadIdx.x + o];
    __syncthreads();
  }
  float rs = rsqrtf(red[0] / (float)kD + 1e-6f);
#pragma unroll
  for (int j = 0; j < 8; ++j) {
    int d = threadIdx.x + 256 * j;
    out[row * kD + d] = (bf16_t)(v[j] * rs * (1.f + scale[d]));
  }
}

// RoPE over one (b,t,head) row; 128 threads = H/2 pairs; writes bf16 (q also scaled)
__global__ void rope_bf16(const float* __restrict__ in, const int* __restrict__ pos,
                          bf16_t* __restrict__ out, int nh, float rscale) {
  long long r = blockIdx.x;          // (b*T + t)*nh + n
  long long bt = r / nh;
  int i = threadIdx.x;               // 0..127
  float p = (float)pos[bt];
  float ts = powf(10000.0f, (2.0f / kH) * (float)i);
  float rad = p / ts;
  float sn = sinf(rad), cs = cosf(rad);
  const float* xr = in + r * kH;
  float x1 = xr[i], x2 = xr[i + kH / 2];
  out[r * kH + i]          = (bf16_t)((x1 * cs - x2 * sn) * rscale);
  out[r * kH + i + kH / 2] = (bf16_t)((x2 * cs + x1 * sn) * rscale);
}

// One block per logits row (b,n,t): causal softmax over s, bf16 probs out
__global__ void softmax_causal_bf16(const float* __restrict__ logits,
                                    bf16_t* __restrict__ probs) {
  __shared__ float red[256];
  long long row = blockIdx.x;        // b*N*T + n*T + t
  int t = (int)(row % kT);
  const float* lr = logits + row * kT;
  bf16_t* pr = probs + row * kT;
  float vals[8];
  float mx = -3.0e38f;
#pragma unroll
  for (int j = 0; j < 8; ++j) {
    int c = threadIdx.x + 256 * j;
    float v = (c <= t) ? lr[c] : -3.0e38f;
    vals[j] = v; mx = fmaxf(mx, v);
  }
  red[threadIdx.x] = mx; __syncthreads();
  for (int o = 128; o > 0; o >>= 1) {
    if ((int)threadIdx.x < o) red[threadIdx.x] = fmaxf(red[threadIdx.x], red[threadIdx.x + o]);
    __syncthreads();
  }
  mx = red[0]; __syncthreads();
  float s = 0.f;
#pragma unroll
  for (int j = 0; j < 8; ++j) {
    int c = threadIdx.x + 256 * j;
    float e = (c <= t) ? expf(vals[j] - mx) : 0.f;
    vals[j] = e; s += e;
  }
  red[threadIdx.x] = s; __syncthreads();
  for (int o = 128; o > 0; o >>= 1) {
    if ((int)threadIdx.x < o) red[threadIdx.x] += red[threadIdx.x + o];
    __syncthreads();
  }
  float inv = 1.0f / red[0];
#pragma unroll
  for (int j = 0; j < 8; ++j) {
    int c = threadIdx.x + 256 * j;
    pr[c] = (bf16_t)(vals[j] * inv);
  }
}

// ---------------------------------------------------------------- WMMA GEMM
// C[M,N] (f32) = A[M,K](bf16) * B (bf16) [+ R]; TRANSB=0: B row-major K x N;
// TRANSB=1: B stored N x K row-major (i.e. B^T). Batched via (b = z/nh,
// n = z%nh) strides. 256 threads = 8 waves; each wave 2x4 16x16 tiles via
// v_wmma_f32_16x16x32_bf16. A-tile (and B-tile when TRANSB=1) staged with
// async Global->LDS b128 copies (ASYNCcnt).
union FragBF { v16bf v; uint4 q[2]; };

template <int TRANSB>
__global__ __launch_bounds__(256) void gemm_bf16(
    const bf16_t* __restrict__ A, const bf16_t* __restrict__ Bm,
    float* __restrict__ C, const float* __restrict__ R,
    int M, int N, int K, int lda, int ldb, int ldc, int nh,
    long long sAb, long long sAn, long long sBb, long long sBn,
    long long sCb, long long sCn) {
  __shared__ bf16_t a_lds[TM * LDSS];
  __shared__ bf16_t b_lds[TN * LDSS];  // stored as Bt[n][k] (k contiguous)
  int z = blockIdx.z;
  int zb = z / nh, zn = z % nh;
  A  += (long long)zb * sAb + (long long)zn * sAn;
  Bm += (long long)zb * sBb + (long long)zn * sBn;
  long long coff = (long long)zb * sCb + (long long)zn * sCn;
  C += coff;
  if (R) R += coff;

  int m0 = blockIdx.y * TM, n0 = blockIdx.x * TN;
  int tid = threadIdx.x;
  int lane = tid & 31, wave = tid >> 5;
  int wr = wave & 3, wc = wave >> 2;       // 4x2 wave grid -> 32x64 per wave
  int l16 = lane & 15, half = lane >> 4;

  // Per-thread staging coordinates
  int srow = tid >> 1, skh = (tid & 1) * 16;            // A (and B when TRANSB=1)
  unsigned aLds = lds_off(&a_lds[srow * LDSS + skh]);
  unsigned bLdsT = lds_off(&b_lds[srow * LDSS + skh]);

  v8f acc[2][4];
#pragma unroll
  for (int i = 0; i < 2; ++i)
#pragma unroll
    for (int j = 0; j < 4; ++j)
#pragma unroll
      for (int r = 0; r < 8; ++r) acc[i][j][r] = 0.0f;

  for (int kt = 0; kt < K; kt += TK) {
    {  // stage A tile 128x32: each thread one 32B chunk via 2 async b128 copies
      const bf16_t* gp = A + (long long)(m0 + srow) * lda + kt + skh;
      async_cp16(aLds, gp);
      async_cp16_off16(aLds, gp);
    }
    if (TRANSB) {   // B stored N x K: k-contiguous per column -> async copies
      const bf16_t* gp = Bm + (long long)(n0 + srow) * ldb + kt + skh;
      async_cp16(bLdsT, gp);
      async_cp16_off16(bLdsT, gp);
    } else {        // B row-major K x N -> transpose into Bt[n][k]
      int kk = tid & 31, nseg = tid >> 5;
      const bf16_t* gp = Bm + (long long)(kt + kk) * ldb + n0 + nseg * 16;
      if (kt + TK < K) __builtin_prefetch(gp + (long long)TK * ldb, 0, 1);
      bf16_t tmp[16];
      const uint4* gq = (const uint4*)gp;
      *(uint4*)&tmp[0] = gq[0]; *(uint4*)&tmp[8] = gq[1];
#pragma unroll
      for (int j = 0; j < 16; ++j) b_lds[(nseg * 16 + j) * LDSS + kk] = tmp[j];
    }
    wait_async0();
    __syncthreads();

    FragBF af[2], bfr[4];
#pragma unroll
    for (int i = 0; i < 2; ++i) {  // A 16x32: lane m=l16; K = half*16 .. +15 contiguous
      int row = wr * 32 + i * 16 + l16;
      af[i].q[0] = *(const uint4*)&a_lds[row * LDSS + half * 16];
      af[i].q[1] = *(const uint4*)&a_lds[row * LDSS + half * 16 + 8];
    }
#pragma unroll
    for (int j = 0; j < 4; ++j) {  // B 32x16: lane n=l16; K = half*16 .. +15 contiguous
      int col = wc * 64 + j * 16 + l16;
      bfr[j].q[0] = *(const uint4*)&b_lds[col * LDSS + half * 16];
      bfr[j].q[1] = *(const uint4*)&b_lds[col * LDSS + half * 16 + 8];
    }
#pragma unroll
    for (int i = 0; i < 2; ++i)
#pragma unroll
      for (int j = 0; j < 4; ++j)
        acc[i][j] = __builtin_amdgcn_wmma_f32_16x16x32_bf16(
            false, af[i].v, false, bfr[j].v, (short)0, acc[i][j], false, false);
    __syncthreads();
  }

  // epilogue: C/D layout -> row = base + r + 8*half, col = base + l16
#pragma unroll
  for (int i = 0; i < 2; ++i)
#pragma unroll
    for (int j = 0; j < 4; ++j) {
      int rbase = m0 + wr * 32 + i * 16 + half * 8;
      int col = n0 + wc * 64 + j * 16 + l16;
#pragma unroll
      for (int r = 0; r < 8; ++r) {
        long long idx = (long long)(rbase + r) * ldc + col;
        float v = acc[i][j][r];
        if (R) v += R[idx];
        C[idx] = v;
      }
    }
}

// Dual-B GEMM fused GeGLU: Out = gelu_tanh(A*B0) * (A*B1), bf16 out. B row-major.
__global__ __launch_bounds__(256) void gemm_dual_geglu(
    const bf16_t* __restrict__ A, const bf16_t* __restrict__ B0,
    const bf16_t* __restrict__ B1, bf16_t* __restrict__ Out,
    int M, int N, int K, int lda, int ldb, int ldo) {
  __shared__ bf16_t a_lds[TM * LDSS];
  __shared__ bf16_t b0_lds[TN * LDSS];
  __shared__ bf16_t b1_lds[TN * LDSS];
  int m0 = blockIdx.y * TM, n0 = blockIdx.x * TN;
  int tid = threadIdx.x;
  int lane = tid & 31, wave = tid >> 5;
  int wr = wave & 3, wc = wave >> 2;
  int l16 = lane & 15, half = lane >> 4;

  int srow = tid >> 1, skh = (tid & 1) * 16;
  unsigned aLds = lds_off(&a_lds[srow * LDSS + skh]);

  v8f accg[2][4], accu[2][4];
#pragma unroll
  for (int i = 0; i < 2; ++i)
#pragma unroll
    for (int j = 0; j < 4; ++j)
#pragma unroll
      for (int r = 0; r < 8; ++r) { accg[i][j][r] = 0.0f; accu[i][j][r] = 0.0f; }

  for (int kt = 0; kt < K; kt += TK) {
    {  // async stage A
      const bf16_t* gp = A + (long long)(m0 + srow) * lda + kt + skh;
      async_cp16(aLds, gp);
      async_cp16_off16(aLds, gp);
    }
    {  // stage both B tiles (transpose through registers)
      int kk = tid & 31, nseg = tid >> 5;
      long long gofs = (long long)(kt + kk) * ldb + n0 + nseg * 16;
      bf16_t tmp[16];
      const uint4* gq = (const uint4*)(B0 + gofs);
      *(uint4*)&tmp[0] = gq[0]; *(uint4*)&tmp[8] = gq[1];
#pragma unroll
      for (int j = 0; j < 16; ++j) b0_lds[(nseg * 16 + j) * LDSS + kk] = tmp[j];
      gq = (const uint4*)(B1 + gofs);
      *(uint4*)&tmp[0] = gq[0]; *(uint4*)&tmp[8] = gq[1];
#pragma unroll
      for (int j = 0; j < 16; ++j) b1_lds[(nseg * 16 + j) * LDSS + kk] = tmp[j];
    }
    wait_async0();
    __syncthreads();

    FragBF af[2], bg[4], bu[4];
#pragma unroll
    for (int i = 0; i < 2; ++i) {
      int row = wr * 32 + i * 16 + l16;
      af[i].q[0] = *(const uint4*)&a_lds[row * LDSS + half * 16];
      af[i].q[1] = *(const uint4*)&a_lds[row * LDSS + half * 16 + 8];
    }
#pragma unroll
    for (int j = 0; j < 4; ++j) {
      int col = wc * 64 + j * 16 + l16;
      bg[j].q[0] = *(const uint4*)&b0_lds[col * LDSS + half * 16];
      bg[j].q[1] = *(const uint4*)&b0_lds[col * LDSS + half * 16 + 8];
      bu[j].q[0] = *(const uint4*)&b1_lds[col * LDSS + half * 16];
      bu[j].q[1] = *(const uint4*)&b1_lds[col * LDSS + half * 16 + 8];
    }
#pragma unroll
    for (int i = 0; i < 2; ++i)
#pragma unroll
      for (int j = 0; j < 4; ++j) {
        accg[i][j] = __builtin_amdgcn_wmma_f32_16x16x32_bf16(
            false, af[i].v, false, bg[j].v, (short)0, accg[i][j], false, false);
        accu[i][j] = __builtin_amdgcn_wmma_f32_16x16x32_bf16(
            false, af[i].v, false, bu[j].v, (short)0, accu[i][j], false, false);
      }
    __syncthreads();
  }

#pragma unroll
  for (int i = 0; i < 2; ++i)
#pragma unroll
    for (int j = 0; j < 4; ++j) {
      int rbase = m0 + wr * 32 + i * 16 + half * 8;
      int col = n0 + wc * 64 + j * 16 + l16;
#pragma unroll
      for (int r = 0; r < 8; ++r) {
        float g = accg[i][j][r], u = accu[i][j][r];
        float t = 0.7978845608028654f * (g + 0.044715f * g * g * g);
        float gl = 0.5f * g * (1.0f + tanhf(t));
        Out[(long long)(rbase + r) * ldo + col] = (bf16_t)(gl * u);
      }
    }
}

// ---------------------------------------------------------------- driver
extern "C" void kernel_launch(void* const* d_in, const int* in_sizes, int n_in,
                              void* d_out, int out_size, void* d_ws, size_t ws_size,
                              hipStream_t stream) {
  (void)in_sizes; (void)n_in; (void)out_size; (void)ws_size;
  const float* x          = (const float*)d_in[0];
  const int*   pos        = (const int*)d_in[1];
  // d_in[2] attn_mask: causal by construction, applied analytically in softmax
  const float* scale_attn = (const float*)d_in[3];
  const float* w_q        = (const float*)d_in[4];
  const float* w_kv       = (const float*)d_in[5];
  const float* w_o        = (const float*)d_in[6];
  const float* scale_ffn  = (const float*)d_in[7];
  const float* w_gating   = (const float*)d_in[8];
  const float* w_linear   = (const float*)d_in[9];
  float* out = (float*)d_out;

  char* base = (char*)d_ws;
  size_t off = 0;
  auto alloc = [&](size_t bytes) -> size_t {
    size_t o = off;
    off = (off + bytes + 255) & ~(size_t)255;
    return o;
  };

  // Persistent region (bf16 weights + x2)
  size_t oWQP = alloc((size_t)kD * kNH * 2);          // (D, N*H) bf16
  size_t oWK  = alloc((size_t)kD * kH * 2);
  size_t oWV  = alloc((size_t)kD * kH * 2);
  size_t oWO  = alloc((size_t)kNH * kD * 2);
  size_t oWG0 = alloc((size_t)kD * kF * 2);
  size_t oWG1 = alloc((size_t)kD * kF * 2);
  size_t oWL  = alloc((size_t)kF * kD * 2);
  size_t oX2  = alloc((size_t)kMT * kD * 4);
  size_t arena = off;

  // Phase A (attention) scratch
  size_t oHBF   = alloc((size_t)kMT * kD * 2);
  size_t oQF    = alloc((size_t)kMT * kNH * 4);
  size_t oKF    = alloc((size_t)kMT * kH * 4);
  size_t oVF    = alloc((size_t)kMT * kH * 4);
  size_t oQBF   = alloc((size_t)kMT * kNH * 2);
  size_t oKBF   = alloc((size_t)kMT * kH * 2);
  size_t oVBF   = alloc((size_t)kMT * kH * 2);
  size_t oLOG   = alloc((size_t)kB * kN * kT * kT * 4);
  size_t oPRB   = alloc((size_t)kB * kN * kT * kT * 2);
  size_t oENC   = alloc((size_t)kMT * kNH * 4);
  size_t oENCBF = alloc((size_t)kMT * kNH * 2);

  // Phase B (FFN) scratch reuses the arena
  off = arena;
  size_t oH2BF = alloc((size_t)kMT * kD * 2);
  size_t oFFBF = alloc((size_t)kMT * kF * 2);

  auto BF = [&](size_t o) { return (bf16_t*)(base + o); };
  auto FP = [&](size_t o) { return (float*)(base + o); };

  auto cvt = [&](const float* in, bf16_t* o, long long n) {
    int blocks = (int)((n + 255) / 256);
    cvt_f32_bf16<<<blocks, 256, 0, stream>>>(in, o, n);
  };
  auto gemm = [&](const bf16_t* A, const bf16_t* Bm, float* C, const float* R,
                  int M, int N, int K, int lda, int ldb, int ldc, int transB,
                  int batch, int nh, long long sAb, long long sAn,
                  long long sBb, long long sBn, long long sCb, long long sCn) {
    dim3 grid(N / TN, M / TM, batch);
    if (transB)
      gemm_bf16<1><<<grid, 256, 0, stream>>>(A, Bm, C, R, M, N, K, lda, ldb, ldc,
                                             nh, sAb, sAn, sBb, sBn, sCb, sCn);
    else
      gemm_bf16<0><<<grid, 256, 0, stream>>>(A, Bm, C, R, M, N, K, lda, ldb, ldc,
                                             nh, sAb, sAn, sBb, sBn, sCb, sCn);
  };

  // ---- weight conversion / packing (bf16)
  cvt(w_kv,                 BF(oWK),  (long long)kD * kH);
  cvt(w_kv + (size_t)kD*kH, BF(oWV),  (long long)kD * kH);
  cvt(w_o,                  BF(oWO),  (long long)kNH * kD);
  cvt(w_gating,                 BF(oWG0), (long long)kD * kF);
  cvt(w_gating + (size_t)kD*kF, BF(oWG1), (long long)kD * kF);
  cvt(w_linear,             BF(oWL),  (long long)kF * kD);
  {
    long long n = (long long)kD * kN * kH;
    pack_wq<<<(int)((n + 255) / 256), 256, 0, stream>>>(w_q, BF(oWQP));
  }

  // ---- h = rmsnorm(x, scale_attn) -> bf16
  rmsnorm_bf16<<<(int)kMT, 256, 0, stream>>>(x, scale_attn, BF(oHBF));

  // ---- projections: q (4096x2048), k/v (4096x256)
  gemm(BF(oHBF), BF(oWQP), FP(oQF), nullptr, (int)kMT, (int)kNH, kD,
       kD, (int)kNH, (int)kNH, 0, 1, 1, 0, 0, 0, 0, 0, 0);
  gemm(BF(oHBF), BF(oWK), FP(oKF), nullptr, (int)kMT, kH, kD,
       kD, kH, kH, 0, 1, 1, 0, 0, 0, 0, 0, 0);
  gemm(BF(oHBF), BF(oWV), FP(oVF), nullptr, (int)kMT, kH, kD,
       kD, kH, kH, 0, 1, 1, 0, 0, 0, 0, 0, 0);

  // ---- RoPE (+ 1/sqrt(H) on q) -> bf16; v -> bf16
  rope_bf16<<<(int)(kMT * kN), 128, 0, stream>>>(FP(oQF), pos, BF(oQBF), kN,
                                                 1.0f / 16.0f);
  rope_bf16<<<(int)kMT, 128, 0, stream>>>(FP(oKF), pos, BF(oKBF), 1, 1.0f);
  cvt(FP(oVF), BF(oVBF), kMT * kH);

  // ---- logits[b,n] = q_bn (T,H) @ k_b^T   (batch = B*N = 16)
  gemm(BF(oQBF), BF(oKBF), FP(oLOG), nullptr, kT, kT, kH,
       (int)kNH, kH, kT, /*transB=*/1, kB * kN, kN,
       (long long)kT * kNH, (long long)kH,          // A: b,n strides
       (long long)kT * kH, 0,                       // B: per-b only
       (long long)kN * kT * kT, (long long)kT * kT);// C: (b,n,t,s)

  // ---- causal softmax -> bf16 probs
  softmax_causal_bf16<<<(int)((long long)kB * kN * kT), 256, 0, stream>>>(
      FP(oLOG), BF(oPRB));

  // ---- enc[b,t,n,:] = probs[b,n] (T,T) @ v_b (T,H)
  gemm(BF(oPRB), BF(oVBF), FP(oENC), nullptr, kT, kH, kT,
       kT, kH, (int)kNH, 0, kB * kN, kN,
       (long long)kN * kT * kT, (long long)kT * kT,
       (long long)kT * kH, 0,
       (long long)kT * kNH, (long long)kH);
  cvt(FP(oENC), BF(oENCBF), kMT * kNH);

  // ---- x2 = enc @ w_o + x
  gemm(BF(oENCBF), BF(oWO), FP(oX2), x, (int)kMT, kD, (int)kNH,
       (int)kNH, kD, kD, 0, 1, 1, 0, 0, 0, 0, 0, 0);

  // ---- h2 = rmsnorm(x2, scale_ffn) -> bf16
  rmsnorm_bf16<<<(int)kMT, 256, 0, stream>>>(FP(oX2), scale_ffn, BF(oH2BF));

  // ---- ff = gelu(h2@Wg0) * (h2@Wg1) -> bf16 (fused dual GEMM)
  {
    dim3 grid(kF / TN, (int)kMT / TM, 1);
    gemm_dual_geglu<<<grid, 256, 0, stream>>>(BF(oH2BF), BF(oWG0), BF(oWG1),
                                              BF(oFFBF), (int)kMT, kF, kD,
                                              kD, kF, kF);
  }

  // ---- out = ff @ w_linear + x2
  gemm(BF(oFFBF), BF(oWL), out, FP(oX2), (int)kMT, kD, kF,
       kF, kD, kD, 0, 1, 1, 0, 0, 0, 0, 0, 0);
}
